// BloqueAttn_19885698581066
// MI455X (gfx1250) — compile-verified, matrix operand order checked by reading, not run on hardware
//
#include <hip/hip_runtime.h>

// ---------------------------------------------------------------------------
// MI455X (gfx1250) fused MHA: projections + RoPE + causal flash-attn + out-proj
// All GEMM math on v_wmma_f32_16x16x32_bf16 (wave32). Global->LDS via CDNA5
// async-to-LDS copies (ASYNCcnt) in a 3-stage pipeline with PARTIAL asynccnt
// waits: each batch gets two compute phases of latency cover and the newest
// batch stays in flight across the barrier. Per K-step all fragments are
// ds-loaded before the WMMA burst (partial dscnt waits, 16 WMMAs per step).
// ---------------------------------------------------------------------------

typedef __bf16 bf16_t;
typedef __attribute__((ext_vector_type(16))) __bf16 v16bf;
typedef __attribute__((ext_vector_type(8)))  float  v8f;

#define L_SEQ   4096
#define D_MODEL 1024
#define NHEAD   16
#define DHEAD   64

static __device__ __forceinline__ bf16_t f2bf(float f) { return (bf16_t)f; }

#define WMMA_BF16(A, B, C) \
  __builtin_amdgcn_wmma_f32_16x16x32_bf16(false, (A), false, (B), (short)0, (C), false, false)

// --- CDNA5 async global->LDS copy (16B per lane), tracked by ASYNCcnt -------
static __device__ __forceinline__ void async_b128(void* lds, const void* g) {
  // Generic pointers to LDS objects carry the wave-relative LDS byte offset in
  // their low 32 bits (ISA 10.2 aperture).
  unsigned int l = (unsigned int)(unsigned long long)(uintptr_t)lds;
  unsigned long long ga = (unsigned long long)(uintptr_t)g;
  asm volatile("global_load_async_to_lds_b128 %0, %1, off"
               :: "v"(l), "v"(ga) : "memory");
}
// Partial/full ASYNCcnt drains (async batches complete in issue order).
#define WAIT_ASYNC(imm) asm volatile("s_wait_asynccnt " #imm ::: "memory")

// ---------------------------------------------------------------------------
// Convert x and the four weight matrices to bf16 (one pass).
// ---------------------------------------------------------------------------
__global__ void conv_inputs(const float* __restrict__ x,
                            const float* __restrict__ Wq, const float* __restrict__ Wk,
                            const float* __restrict__ Wv, const float* __restrict__ Wo,
                            bf16_t* __restrict__ xb,
                            bf16_t* __restrict__ wqb, bf16_t* __restrict__ wkb,
                            bf16_t* __restrict__ wvb, bf16_t* __restrict__ wob) {
  int i = blockIdx.x * blockDim.x + threadIdx.x;
  const int nx = L_SEQ * D_MODEL;
  const int nw = D_MODEL * D_MODEL;
  if (i < nx) xb[i] = f2bf(x[i]);
  if (i < nw) {
    wqb[i] = f2bf(Wq[i]);
    wkb[i] = f2bf(Wk[i]);
    wvb[i] = f2bf(Wv[i]);
    wob[i] = f2bf(Wo[i]);
  }
}

// ---------------------------------------------------------------------------
// GEMM: C[M,N] (f32) = A[M,K] bf16 row-major x W^T, W[N,K] bf16 row-major.
// Block = 256 threads = 8 waves; block tile 256(M) x 64(N); K-step 64.
// Each wave: 32x64 tile = 2 row-groups x 4 col-tiles (8 f32 accumulators).
// A/B tiles triple-buffered in LDS via async-to-LDS (batch = 10 asyncs/thread
// -> steady-state s_wait_asynccnt 0xa). Fragment k order used here (contiguous, not ISA table order):
// A 16x32 lane=(k/16)*16+m; B 32x16 lane=(k/16)*16+n, 16 contiguous bf16.
// ---------------------------------------------------------------------------
__global__ void __launch_bounds__(256)
gemm_bf16(const bf16_t* __restrict__ A, const bf16_t* __restrict__ W,
          float* __restrict__ C, int M, int N, int K) {
  __shared__ bf16_t a_sh[3][256 * 64];   // 3 x 32 KB
  __shared__ bf16_t b_sh[3][64 * 64];    // 3 x  8 KB

  const int tid  = threadIdx.x;
  const int wave = tid >> 5;
  const int lane = tid & 31;
  const int m_in = lane & 15;
  const int kh   = lane >> 4;
  const int m_base = blockIdx.y * 256;
  const int n0     = blockIdx.x * 64;

  // Stage one K-step: A tile 256x64 (2048 16B-chunks), B tile 64x64 (512).
  // 10 async copies per thread = one ASYNCcnt batch.
  auto stage = [&](int buf, int k0) {
#pragma unroll
    for (int i = 0; i < 8; ++i) {
      int c = tid + i * 256;
      int row = c >> 3, col = (c & 7) * 8;
      async_b128(&a_sh[buf][row * 64 + col],
                 A + (size_t)(m_base + row) * K + k0 + col);
    }
#pragma unroll
    for (int i = 0; i < 2; ++i) {
      int c = tid + i * 256;
      int row = c >> 3, col = (c & 7) * 8;
      async_b128(&b_sh[buf][row * 64 + col],
                 W + (size_t)(n0 + row) * K + k0 + col);
    }
  };

  v8f acc[2][4] = {};

  stage(0, 0);
  stage(1, 64);
  WAIT_ASYNC(0xa);       // batch 0 landed; batch 1 still in flight
  __syncthreads();

  int buf = 0;
  for (int k0 = 0; k0 < K; k0 += 64) {
    const int nxt = k0 + 128;
    const bool issued = (nxt < K);
    int sbuf = buf + 2; if (sbuf >= 3) sbuf -= 3;
    if (issued) stage(sbuf, nxt);  // 2-ahead prefetch into the retired buffer

    const bf16_t* as = &a_sh[buf][(wave * 32 + m_in) * 64 + kh * 16];
    const bf16_t* bs = &b_sh[buf][m_in * 64 + kh * 16];

    // Load ALL fragments first, then a back-to-back WMMA burst.
    v16bf a[2][2], b[2][4];
#pragma unroll
    for (int g = 0; g < 2; ++g)
#pragma unroll
      for (int kk = 0; kk < 2; ++kk)
        a[g][kk] = *(const v16bf*)(as + g * 16 * 64 + kk * 32);
#pragma unroll
    for (int kk = 0; kk < 2; ++kk)
#pragma unroll
      for (int t = 0; t < 4; ++t)
        b[kk][t] = *(const v16bf*)(bs + kk * 32 + t * 16 * 64);

#pragma unroll
    for (int kk = 0; kk < 2; ++kk)
#pragma unroll
      for (int g = 0; g < 2; ++g)
#pragma unroll
        for (int t = 0; t < 4; ++t)
          acc[g][t] = WMMA_BF16(a[g][kk], b[kk][t], acc[g][t]);

    // Guarantee the NEXT buffer's batch arrived; keep the newest in flight.
    if (issued) { WAIT_ASYNC(0xa); } else { WAIT_ASYNC(0x0); }
    __syncthreads();
    buf += 1; if (buf >= 3) buf -= 3;
  }

  // C/D layout: VGPR i, lanes 0-15 -> M=i, lanes 16-31 -> M=i+8; N = lane&15.
  const int rb = m_base + wave * 32 + kh * 8;
#pragma unroll
  for (int g = 0; g < 2; ++g)
#pragma unroll
    for (int i = 0; i < 8; ++i) {
      float* cr = C + (size_t)(rb + g * 16 + i) * N + n0 + m_in;
      cr[ 0] = acc[g][0][i];
      cr[16] = acc[g][1][i];
      cr[32] = acc[g][2][i];
      cr[48] = acc[g][3][i];
    }
}

// ---------------------------------------------------------------------------
// RoPE + f32->bf16. out[p] = x[2p]*cos - x[2p+1]*sin (p < 32),
//                   out[32+p] = x[2p]*sin + x[2p+1]*cos.
// ---------------------------------------------------------------------------
__global__ void rope_to_bf16(const float* __restrict__ src, bf16_t* __restrict__ dst) {
  int idx = blockIdx.x * blockDim.x + threadIdx.x;
  if (idx >= L_SEQ * NHEAD * 32) return;
  int p = idx & 31;
  int h = (idx >> 5) & (NHEAD - 1);
  int l = idx >> 9;
  float inv = __expf(-(float)(2 * p) * (9.210340371976184f / 64.0f)); // ln(1e4)
  float ang = (float)l * inv;
  float c = cosf(ang), s = sinf(ang);
  size_t base = (size_t)l * D_MODEL + h * DHEAD;
  float x1 = src[base + 2 * p];
  float x2 = src[base + 2 * p + 1];
  dst[base + p]      = f2bf(x1 * c - x2 * s);
  dst[base + 32 + p] = f2bf(x1 * s + x2 * c);
}

// ---------------------------------------------------------------------------
// V (f32 [L,D]) -> Vt (bf16 [D,L]) so P.V B-fragments are contiguous per lane.
// ---------------------------------------------------------------------------
__global__ void vt_conv(const float* __restrict__ Vf, bf16_t* __restrict__ Vt) {
  int idx = blockIdx.x * blockDim.x + threadIdx.x;
  if (idx >= L_SEQ * D_MODEL) return;
  int d = idx & (D_MODEL - 1);
  int l = idx >> 10;
  Vt[(size_t)d * L_SEQ + l] = f2bf(Vf[idx]);
}

// ---------------------------------------------------------------------------
// Causal flash attention. Grid = (L/128, H). Block = 128 threads = 4 waves;
// each wave owns 32 query rows (2 row-groups). K/V 64x64 tiles triple-buffered
// in LDS via async copies (batch = 8 asyncs/thread -> s_wait_asynccnt 0x8),
// 2-ahead prefetch. V fragments load before the softmax so exp/shfl VALU work
// hides their LDS latency; only P fragments wait on p_lds stores.
// ---------------------------------------------------------------------------
__global__ void __launch_bounds__(128)
flash_attn(const bf16_t* __restrict__ Q, const bf16_t* __restrict__ Kmat,
           const bf16_t* __restrict__ Vt, bf16_t* __restrict__ O) {
  __shared__ bf16_t k_sh[3][64 * 64];     // 3 x 8 KB: K rows x dh
  __shared__ bf16_t v_sh[3][64 * 64];     // 3 x 8 KB: dh x key (from Vt)
  __shared__ bf16_t p_lds[4 * 32 * 64];   // 16 KB: per-wave P tiles (32x64)

  const int tid  = threadIdx.x;
  const int wave = tid >> 5;
  const int lane = tid & 31;
  const int m_in = lane & 15;
  const int kh   = lane >> 4;
  const int h    = blockIdx.y;
  const int qb   = blockIdx.x;
  const int q0w  = qb * 128 + wave * 32;  // first query row of this wave

  // Stage key-block kb: K tile + V tile (8 async copies per thread = 1 batch).
  auto stage = [&](int buf, int kb) {
#pragma unroll
    for (int i = 0; i < 4; ++i) {
      int c = tid + i * 128;
      int row = c >> 3, col = (c & 7) * 8;
      async_b128(&k_sh[buf][row * 64 + col],
                 Kmat + (size_t)(kb * 64 + row) * D_MODEL + h * DHEAD + col);
      async_b128(&v_sh[buf][row * 64 + col],
                 Vt + (size_t)(h * DHEAD + row) * L_SEQ + kb * 64 + col);
    }
  };

  // Q fragments: 2 row-groups x dh {0..31, 32..63}.
  v16bf qa[2][2];
#pragma unroll
  for (int g = 0; g < 2; ++g) {
    const bf16_t* qrow = Q + (size_t)(q0w + g * 16 + m_in) * D_MODEL + h * DHEAD + kh * 16;
    qa[g][0] = *(const v16bf*)(qrow);
    qa[g][1] = *(const v16bf*)(qrow + 32);
  }

  v8f o[2][4] = {};
  float rmax[2][8], rsum[2][8];
#pragma unroll
  for (int g = 0; g < 2; ++g)
#pragma unroll
    for (int i = 0; i < 8; ++i) { rmax[g][i] = -__builtin_inff(); rsum[g][i] = 0.0f; }

  bf16_t* pl = p_lds + wave * 32 * 64;
  const float scale = 0.125f;     // DH^-0.5
  const int kb_max = 2 * qb + 1;  // last key block for this 128-query tile

  stage(0, 0);
  stage(1, 1);                    // kb_max >= 1 always
  WAIT_ASYNC(0x8);                // batch 0 landed; batch 1 in flight
  __syncthreads();

  int buf = 0;
  for (int kb = 0; kb <= kb_max; ++kb) {
    const bool issued = (kb + 2 <= kb_max);
    int sbuf = buf + 2; if (sbuf >= 3) sbuf -= 3;
    if (issued) stage(sbuf, kb + 2);   // 2-ahead prefetch

    // ---- S = Q x K^T (32x64): load all K fragments, then WMMA burst ----
    const bf16_t* ks = &k_sh[buf][m_in * 64 + kh * 16];
    v16bf kf[2][4];
#pragma unroll
    for (int kk = 0; kk < 2; ++kk)
#pragma unroll
      for (int t = 0; t < 4; ++t)
        kf[kk][t] = *(const v16bf*)(ks + kk * 32 + t * 16 * 64);

    v8f s[2][4] = {};
#pragma unroll
    for (int kk = 0; kk < 2; ++kk)
#pragma unroll
      for (int g = 0; g < 2; ++g)
#pragma unroll
        for (int t = 0; t < 4; ++t)
          s[g][t] = WMMA_BF16(qa[g][kk], kf[kk][t], s[g][t]);

    // ---- V fragments early: their LDS latency hides under softmax ------
    const bf16_t* vs = &v_sh[buf][m_in * 64 + kh * 16];
    v16bf vf[2][4];
#pragma unroll
    for (int kk = 0; kk < 2; ++kk)
#pragma unroll
      for (int t = 0; t < 4; ++t)
        vf[kk][t] = *(const v16bf*)(vs + kk * 32 + t * 16 * 64);

    // ---- online softmax ------------------------------------------------
    const bool needMask = (kb * 64 + 63) > q0w;   // block reaches the diagonal
#pragma unroll
    for (int g = 0; g < 2; ++g)
#pragma unroll
      for (int i = 0; i < 8; ++i) {
        const int qg = q0w + g * 16 + kh * 8 + i;  // global query row
        const int kg = kb * 64 + m_in;             // global key col (tile 0)
        float v0 = s[g][0][i] * scale; if (needMask && (kg      > qg)) v0 = -__builtin_inff();
        float v1 = s[g][1][i] * scale; if (needMask && (kg + 16 > qg)) v1 = -__builtin_inff();
        float v2 = s[g][2][i] * scale; if (needMask && (kg + 32 > qg)) v2 = -__builtin_inff();
        float v3 = s[g][3][i] * scale; if (needMask && (kg + 48 > qg)) v3 = -__builtin_inff();

        float bm = fmaxf(fmaxf(v0, v1), fmaxf(v2, v3));
        bm = fmaxf(bm, __shfl_xor(bm, 1, 32));
        bm = fmaxf(bm, __shfl_xor(bm, 2, 32));
        bm = fmaxf(bm, __shfl_xor(bm, 4, 32));
        bm = fmaxf(bm, __shfl_xor(bm, 8, 32));

        float mnew = fmaxf(rmax[g][i], bm);
        float sc   = __expf(rmax[g][i] - mnew);
        rmax[g][i] = mnew;

        v0 = __expf(v0 - mnew);
        v1 = __expf(v1 - mnew);
        v2 = __expf(v2 - mnew);
        v3 = __expf(v3 - mnew);

        float ps = v0 + v1 + v2 + v3;
        ps += __shfl_xor(ps, 1, 32);
        ps += __shfl_xor(ps, 2, 32);
        ps += __shfl_xor(ps, 4, 32);
        ps += __shfl_xor(ps, 8, 32);
        rsum[g][i] = rsum[g][i] * sc + ps;

        o[g][0][i] *= sc; o[g][1][i] *= sc; o[g][2][i] *= sc; o[g][3][i] *= sc;

        // stage P (bf16) row-major [32 x 64] for A-fragment reload
        const int r = g * 16 + kh * 8 + i;
        pl[r * 64 +      m_in] = f2bf(v0);
        pl[r * 64 + 16 + m_in] = f2bf(v1);
        pl[r * 64 + 32 + m_in] = f2bf(v2);
        pl[r * 64 + 48 + m_in] = f2bf(v3);
      }

    // ---- O += P x V ----------------------------------------------------
    v16bf pa[2][2];
#pragma unroll
    for (int g = 0; g < 2; ++g)
#pragma unroll
      for (int kk = 0; kk < 2; ++kk)
        pa[g][kk] = *(const v16bf*)(pl + (g * 16 + m_in) * 64 + kk * 32 + kh * 16);

#pragma unroll
    for (int kk = 0; kk < 2; ++kk)
#pragma unroll
      for (int g = 0; g < 2; ++g)
#pragma unroll
        for (int t = 0; t < 4; ++t)
          o[g][t] = WMMA_BF16(pa[g][kk], vf[kk][t], o[g][t]);

    if (issued) { WAIT_ASYNC(0x8); } else { WAIT_ASYNC(0x0); }
    __syncthreads();
    buf += 1; if (buf >= 3) buf -= 3;
  }

  // ---- normalize and store O (bf16, [L, D] head-major columns) ---------
#pragma unroll
  for (int g = 0; g < 2; ++g)
#pragma unroll
    for (int i = 0; i < 8; ++i) {
      float inv = 1.0f / rsum[g][i];
      int row = q0w + g * 16 + kh * 8 + i;
      bf16_t* orow = O + (size_t)row * D_MODEL + h * DHEAD + m_in;
      orow[ 0] = f2bf(o[g][0][i] * inv);
      orow[16] = f2bf(o[g][1][i] * inv);
      orow[32] = f2bf(o[g][2][i] * inv);
      orow[48] = f2bf(o[g][3][i] * inv);
    }
}

// ---------------------------------------------------------------------------
// Host launch. d_out (16 MB f32) doubles as the fp32 scratch for pre-RoPE
// projections; workspace usage = 48 MB of bf16 buffers.
// ---------------------------------------------------------------------------
extern "C" void kernel_launch(void* const* d_in, const int* in_sizes, int n_in,
                              void* d_out, int out_size, void* d_ws, size_t ws_size,
                              hipStream_t stream) {
  const float* x  = (const float*)d_in[0];
  // d_in[1] = causal mask (unused; causality is structural in flash_attn)
  const float* Wq = (const float*)d_in[2];
  const float* Wk = (const float*)d_in[3];
  const float* Wv = (const float*)d_in[4];
  const float* Wo = (const float*)d_in[5];

  char* ws = (char*)d_ws;
  const size_t MB = 1u << 20;
  bf16_t* xb  = (bf16_t*)(ws +  0 * MB);  //  8 MB: x in bf16
  bf16_t* wqb = (bf16_t*)(ws +  8 * MB);  //  2 MB
  bf16_t* wkb = (bf16_t*)(ws + 10 * MB);  //  2 MB
  bf16_t* wvb = (bf16_t*)(ws + 12 * MB);  //  2 MB
  bf16_t* wob = (bf16_t*)(ws + 14 * MB);  //  2 MB
  bf16_t* Qb  = (bf16_t*)(ws + 16 * MB);  //  8 MB: RoPE'd Q
  bf16_t* Kbb = (bf16_t*)(ws + 24 * MB);  //  8 MB: RoPE'd K
  bf16_t* Vt  = (bf16_t*)(ws + 32 * MB);  //  8 MB: V transposed [D,L]
  bf16_t* Ob  = (bf16_t*)(ws + 40 * MB);  //  8 MB: attention output

  float* tmpF = (float*)d_out;            // fp32 staging (overwritten last)

  const int nElem = L_SEQ * D_MODEL;
  dim3 gGemm(D_MODEL / 64, L_SEQ / 256);
  dim3 bGemm(256);

  conv_inputs<<<(nElem + 255) / 256, 256, 0, stream>>>(x, Wq, Wk, Wv, Wo,
                                                       xb, wqb, wkb, wvb, wob);

  // Q = x Wq^T -> RoPE -> bf16
  gemm_bf16<<<gGemm, bGemm, 0, stream>>>(xb, wqb, tmpF, L_SEQ, D_MODEL, D_MODEL);
  rope_to_bf16<<<(L_SEQ * NHEAD * 32 + 255) / 256, 256, 0, stream>>>(tmpF, Qb);

  // K = x Wk^T -> RoPE -> bf16
  gemm_bf16<<<gGemm, bGemm, 0, stream>>>(xb, wkb, tmpF, L_SEQ, D_MODEL, D_MODEL);
  rope_to_bf16<<<(L_SEQ * NHEAD * 32 + 255) / 256, 256, 0, stream>>>(tmpF, Kbb);

  // V = x Wv^T -> transpose -> bf16 [D,L]
  gemm_bf16<<<gGemm, bGemm, 0, stream>>>(xb, wvb, tmpF, L_SEQ, D_MODEL, D_MODEL);
  vt_conv<<<(nElem + 255) / 256, 256, 0, stream>>>(tmpF, Vt);

  // causal flash attention
  flash_attn<<<dim3(L_SEQ / 128, NHEAD), 128, 0, stream>>>(Qb, Kbb, Vt, Ob);

  // out = O Wo^T (fp32 into d_out)
  gemm_bf16<<<gGemm, bGemm, 0, stream>>>(Ob, wob, (float*)d_out, L_SEQ, D_MODEL, D_MODEL);
}
